// MultiHeadAttention_3418793967685
// MI455X (gfx1250) — matrix operand order repaired and checked by reading, not compiled, not run on hardware
//
#include <hip/hip_runtime.h>
#include <hip/hip_bf16.h>

typedef __bf16 bf16_t;
typedef __attribute__((ext_vector_type(4)))  bf16_t v4bf;
typedef __attribute__((ext_vector_type(8)))  bf16_t v8bf;
typedef __attribute__((ext_vector_type(16))) bf16_t v16bf;
typedef __attribute__((ext_vector_type(8)))  float  v8f;

#define B_  8
#define S_  1024
#define E_  768
#define H_  12
#define E3_ (3 * E_)
#define HE_ (H_ * E_)

// ---------------------------------------------------------------------------
// WMMA fragment helpers (wave32, v_wmma_f32_16x16x32_bf16)
// A (16x32, MxK): lane L holds row M=L&15; K = k0+16*(L>>4)+[0..15] contiguous
//                 (matches B order)
// B (32x16, KxN): lane L holds col N=L&15; K = k0+16*(L>>4)+[0..15] contiguous
// C/D (16x16 f32): lane L, vgpr v -> M = v + 8*(L>>4), N = L&15
// ---------------------------------------------------------------------------

__device__ __forceinline__ v16bf load_fragA(const bf16_t* __restrict__ base, int ld,
                                            int m0, int k0) {
  const int lane = (int)(threadIdx.x & 31u);
  const bf16_t* p = base + (size_t)(m0 + (lane & 15)) * ld + (k0 + ((lane >> 4) << 4));
  v8bf lo = *(const v8bf*)(p);
  v8bf hi = *(const v8bf*)(p + 8);
  v16bf r;
#pragma unroll
  for (int i = 0; i < 8; ++i) { r[i] = lo[i]; r[i + 8] = hi[i]; }
  return r;
}

// source stored so element (k, n) lives at base[(n0+n)*ld + k]  (row-major [N][K])
__device__ __forceinline__ v16bf load_fragB(const bf16_t* __restrict__ base, int ld,
                                            int n0, int k0) {
  const int lane = (int)(threadIdx.x & 31u);
  const bf16_t* p = base + (size_t)(n0 + (lane & 15)) * ld + (k0 + ((lane >> 4) << 4));
  return *(const v16bf*)p;
}

__device__ __forceinline__ v8f wmma_bf16(v16bf a, v16bf b, v8f c) {
  return __builtin_amdgcn_wmma_f32_16x16x32_bf16(false, a, false, b, (short)0, c,
                                                 false, false);
}

// 64x64 output tile per wave: 4 M-tiles x 4 N-tiles = 16 accumulators
// (128 VGPRs); 16 WMMA per 32-wide k step, 8 fragment loads per k step.
// Requires __launch_bounds__(256, 1) on callers so the allocator grants
// ~192 VGPRs instead of spilling at the 8-wave occupancy cap.
__device__ __forceinline__ void gemm64x64(const bf16_t* __restrict__ A, int lda, int m0,
                                          const bf16_t* __restrict__ Bm, int ldb, int n0,
                                          int K, v8f acc[4][4]) {
  for (int k0 = 0; k0 < K; k0 += 32) {
    v16bf a[4];
#pragma unroll
    for (int i = 0; i < 4; ++i) a[i] = load_fragA(A, lda, m0 + 16 * i, k0);
#pragma unroll
    for (int j = 0; j < 4; ++j) {
      v16bf b = load_fragB(Bm, ldb, n0 + 16 * j, k0);
#pragma unroll
      for (int i = 0; i < 4; ++i) acc[i][j] = wmma_bf16(a[i], b, acc[i][j]);
    }
  }
}

__device__ __forceinline__ void zero_acc(v8f acc[4][4]) {
#pragma unroll
  for (int i = 0; i < 4; ++i)
#pragma unroll
    for (int j = 0; j < 4; ++j)
      acc[i][j] = (v8f){0.f, 0.f, 0.f, 0.f, 0.f, 0.f, 0.f, 0.f};
}

// ---------------------------------------------------------------------------
// Kernels
// ---------------------------------------------------------------------------

__global__ void __launch_bounds__(256) mha_cvt_bf16(const float* __restrict__ src,
                                                    bf16_t* __restrict__ dst, int n) {
  int i = (int)(blockIdx.x * 256u + threadIdx.x);
  if (i < n) dst[i] = (bf16_t)src[i];
}

// QKV projection: per (b,h): [S,E] x [E,3E] + bias.  Q scaled by 1/sqrt(E),
// Q/K stored [S,E] bf16, V stored transposed [E,S] bf16.
__global__ void __launch_bounds__(256, 1)
mha_qkv(const bf16_t* __restrict__ xh, const bf16_t* __restrict__ wh,
        const float* __restrict__ qkv_b, bf16_t* __restrict__ Q,
        bf16_t* __restrict__ Km, bf16_t* __restrict__ Vt) {
  const int NT = E3_ / 64;  // 36
  const int MT = S_ / 64;   // 16
  int wid = (int)((blockIdx.x * 256u + threadIdx.x) >> 5);
  int nt = wid % NT;
  int t = wid / NT;
  int mt = t % MT;
  int bh = t / MT;
  if (bh >= B_ * H_) return;
  int b = bh / H_, h = bh % H_;

  const bf16_t* A = xh + (size_t)b * S_ * E_;
  const bf16_t* W = wh + (size_t)h * E3_ * E_;
  const int m0 = mt * 64, n0 = nt * 64;

  v8f acc[4][4];
  zero_acc(acc);
  gemm64x64(A, E_, m0, W, E_, n0, E_, acc);

  const int lane = (int)(threadIdx.x & 31u);
  const float qscale = 0.03608439182435161f;  // 1/sqrt(768)

#pragma unroll
  for (int j = 0; j < 4; ++j) {
    const int f0 = n0 + 16 * j;
    const float bias = qkv_b[h * E3_ + f0 + (lane & 15)];
    if (f0 < E_) {  // Q (scaled)
      bf16_t* dst = Q + (size_t)bh * S_ * E_;
      const int col = f0 + (lane & 15);
#pragma unroll
      for (int i = 0; i < 4; ++i) {
        const int rb = m0 + 16 * i + ((lane >> 4) << 3);
#pragma unroll
        for (int v = 0; v < 8; ++v)
          dst[(size_t)(rb + v) * E_ + col] = (bf16_t)((acc[i][j][v] + bias) * qscale);
      }
    } else if (f0 < 2 * E_) {  // K
      bf16_t* dst = Km + (size_t)bh * S_ * E_;
      const int col = f0 - E_ + (lane & 15);
#pragma unroll
      for (int i = 0; i < 4; ++i) {
        const int rb = m0 + 16 * i + ((lane >> 4) << 3);
#pragma unroll
        for (int v = 0; v < 8; ++v)
          dst[(size_t)(rb + v) * E_ + col] = (bf16_t)(acc[i][j][v] + bias);
      }
    } else {  // V transposed: Vt[e][s] -- per-lane 8 consecutive s -> vector store
      bf16_t* dst = Vt + (size_t)bh * E_ * S_;
      const int e = f0 - 2 * E_ + (lane & 15);
#pragma unroll
      for (int i = 0; i < 4; ++i) {
        const int rb = m0 + 16 * i + ((lane >> 4) << 3);
        v8bf pack;
#pragma unroll
        for (int v = 0; v < 8; ++v) pack[v] = (bf16_t)(acc[i][j][v] + bias);
        *(v8bf*)(dst + (size_t)e * S_ + rb) = pack;
      }
    }
  }
}

// scores = Q * K^T  (scale already folded into Q), fp32 out [B,H,S,S]
__global__ void __launch_bounds__(256, 1)
mha_scores(const bf16_t* __restrict__ Q, const bf16_t* __restrict__ Km,
           float* __restrict__ Sc) {
  const int NT = S_ / 64;  // 16
  const int MT = S_ / 64;  // 16
  int wid = (int)((blockIdx.x * 256u + threadIdx.x) >> 5);
  int nt = wid % NT;
  int t = wid / NT;
  int mt = t % MT;
  int bh = t / MT;
  if (bh >= B_ * H_) return;

  const bf16_t* A = Q + (size_t)bh * S_ * E_;
  const bf16_t* Bm = Km + (size_t)bh * S_ * E_;
  const int m0 = mt * 64, n0 = nt * 64;

  v8f acc[4][4];
  zero_acc(acc);
  gemm64x64(A, E_, m0, Bm, E_, n0, E_, acc);

  float* dst = Sc + (size_t)bh * S_ * S_;
  const int lane = (int)(threadIdx.x & 31u);
#pragma unroll
  for (int j = 0; j < 4; ++j) {
    const int col = n0 + 16 * j + (lane & 15);
#pragma unroll
    for (int i = 0; i < 4; ++i) {
      const int rb = m0 + 16 * i + ((lane >> 4) << 3);
#pragma unroll
      for (int v = 0; v < 8; ++v) dst[(size_t)(rb + v) * S_ + col] = acc[i][j][v];
    }
  }
}

// row softmax over 1024 columns; one wave per row; bf16 probabilities out
__global__ void __launch_bounds__(256) mha_softmax(const float* __restrict__ Sc,
                                                   bf16_t* __restrict__ P) {
  const int rows = B_ * H_ * S_;
  int wid = (int)((blockIdx.x * 256u + threadIdx.x) >> 5);
  if (wid >= rows) return;
  const int lane = (int)(threadIdx.x & 31u);
  const float4* row = (const float4*)(Sc + (size_t)wid * S_);
  v4bf* prow = (v4bf*)(P + (size_t)wid * S_);

  float4 v[8];
  float m = -3.402823466e38f;
#pragma unroll
  for (int i = 0; i < 8; ++i) {
    v[i] = row[lane + 32 * i];
    m = fmaxf(m, fmaxf(fmaxf(v[i].x, v[i].y), fmaxf(v[i].z, v[i].w)));
  }
#pragma unroll
  for (int off = 16; off >= 1; off >>= 1) m = fmaxf(m, __shfl_xor(m, off, 32));

  float s = 0.f;
#pragma unroll
  for (int i = 0; i < 8; ++i) {
    v[i].x = __expf(v[i].x - m);
    v[i].y = __expf(v[i].y - m);
    v[i].z = __expf(v[i].z - m);
    v[i].w = __expf(v[i].w - m);
    s += (v[i].x + v[i].y) + (v[i].z + v[i].w);
  }
#pragma unroll
  for (int off = 16; off >= 1; off >>= 1) s += __shfl_xor(s, off, 32);
  const float inv = 1.0f / s;

#pragma unroll
  for (int i = 0; i < 8; ++i) {
    v4bf o;
    o[0] = (bf16_t)(v[i].x * inv);
    o[1] = (bf16_t)(v[i].y * inv);
    o[2] = (bf16_t)(v[i].z * inv);
    o[3] = (bf16_t)(v[i].w * inv);
    prow[lane + 32 * i] = o;
  }
}

// O = P * V  (V stored transposed [E,S] -> contiguous B fragments).
// Written directly into concat layout Oc[B,S,H*E] bf16.
__global__ void __launch_bounds__(256, 1)
mha_pv(const bf16_t* __restrict__ P, const bf16_t* __restrict__ Vt,
       bf16_t* __restrict__ Oc) {
  const int NT = E_ / 64;  // 12
  const int MT = S_ / 64;  // 16
  int wid = (int)((blockIdx.x * 256u + threadIdx.x) >> 5);
  int nt = wid % NT;
  int t = wid / NT;
  int mt = t % MT;
  int bh = t / MT;
  if (bh >= B_ * H_) return;
  int b = bh / H_, h = bh % H_;

  const bf16_t* A = P + (size_t)bh * S_ * S_;
  const bf16_t* Bm = Vt + (size_t)bh * E_ * S_;
  const int m0 = mt * 64, n0 = nt * 64;

  v8f acc[4][4];
  zero_acc(acc);
  gemm64x64(A, S_, m0, Bm, S_, n0, S_, acc);

  bf16_t* dst = Oc + (size_t)b * S_ * HE_;
  const int lane = (int)(threadIdx.x & 31u);
#pragma unroll
  for (int j = 0; j < 4; ++j) {
    const int col = h * E_ + n0 + 16 * j + (lane & 15);
#pragma unroll
    for (int i = 0; i < 4; ++i) {
      const int rb = m0 + 16 * i + ((lane >> 4) << 3);
#pragma unroll
      for (int v = 0; v < 8; ++v)
        dst[(size_t)(rb + v) * HE_ + col] = (bf16_t)acc[i][j][v];
    }
  }
}

// final projection: [B*S, H*E] x [H*E, E] + out_b -> fp32 d_out
__global__ void __launch_bounds__(256, 1)
mha_out(const bf16_t* __restrict__ Oc, const bf16_t* __restrict__ Wo,
        const float* __restrict__ out_b, float* __restrict__ out) {
  const int NT = E_ / 64;         // 12
  const int MT = (B_ * S_) / 64;  // 128
  int wid = (int)((blockIdx.x * 256u + threadIdx.x) >> 5);
  int nt = wid % NT;
  int mt = wid / NT;
  if (mt >= MT) return;

  const int m0 = mt * 64, n0 = nt * 64;
  v8f acc[4][4];
  zero_acc(acc);
  gemm64x64(Oc, HE_, m0, Wo, HE_, n0, HE_, acc);

  const int lane = (int)(threadIdx.x & 31u);
#pragma unroll
  for (int j = 0; j < 4; ++j) {
    const int col = n0 + 16 * j + (lane & 15);
    const float bias = out_b[col];
#pragma unroll
    for (int i = 0; i < 4; ++i) {
      const int rb = m0 + 16 * i + ((lane >> 4) << 3);
#pragma unroll
      for (int v = 0; v < 8; ++v)
        out[(size_t)(rb + v) * E_ + col] = acc[i][j][v] + bias;
    }
  }
}

// ---------------------------------------------------------------------------

extern "C" void kernel_launch(void* const* d_in, const int* in_sizes, int n_in,
                              void* d_out, int out_size, void* d_ws, size_t ws_size,
                              hipStream_t stream) {
  const float* x      = (const float*)d_in[0];  // [B,S,E]
  const float* qkv_w  = (const float*)d_in[1];  // [H,3E,E]
  const float* qkv_b  = (const float*)d_in[2];  // [H,3E]
  const float* out_w  = (const float*)d_in[3];  // [E,H*E]
  const float* out_b  = (const float*)d_in[4];  // [E]
  float* out = (float*)d_out;                   // [B,S,E]

  size_t off = 0;
  auto carve = [&](size_t bytes) -> void* {
    void* p = (char*)d_ws + off;
    off += (bytes + 255) & ~(size_t)255;
    return p;
  };
  const size_t nx = (size_t)B_ * S_ * E_;        // 6.29M
  const size_t nw = (size_t)H_ * E3_ * E_;       // 21.2M
  const size_t no = (size_t)E_ * HE_;            // 7.08M
  const size_t nqkv = (size_t)B_ * H_ * S_ * E_; // 75.5M
  const size_t nsc = (size_t)B_ * H_ * S_ * S_;  // 100.7M

  bf16_t* xh  = (bf16_t*)carve(nx * 2);
  bf16_t* wh  = (bf16_t*)carve(nw * 2);
  bf16_t* woh = (bf16_t*)carve(no * 2);
  bf16_t* Q   = (bf16_t*)carve(nqkv * 2);
  bf16_t* Km  = (bf16_t*)carve(nqkv * 2);
  bf16_t* Vt  = (bf16_t*)carve(nqkv * 2);
  float*  Sc  = (float*)carve(nsc * 4);
  bf16_t* P   = (bf16_t*)carve(nsc * 2);
  bf16_t* Oc  = (bf16_t*)carve(nqkv * 2);

  mha_cvt_bf16<<<(unsigned)((nx + 255) / 256), 256, 0, stream>>>(x, xh, (int)nx);
  mha_cvt_bf16<<<(unsigned)((nw + 255) / 256), 256, 0, stream>>>(qkv_w, wh, (int)nw);
  mha_cvt_bf16<<<(unsigned)((no + 255) / 256), 256, 0, stream>>>(out_w, woh, (int)no);

  // waves: 96*16*36 = 55296 -> 6912 blocks of 8 waves
  mha_qkv<<<6912, 256, 0, stream>>>(xh, wh, qkv_b, Q, Km, Vt);
  // waves: 96*16*16 = 24576 -> 3072 blocks
  mha_scores<<<3072, 256, 0, stream>>>(Q, Km, Sc);
  // rows: 96*1024 = 98304 -> 12288 blocks of 8 waves
  mha_softmax<<<12288, 256, 0, stream>>>(Sc, P);
  // waves: 96*16*12 = 18432 -> 2304 blocks
  mha_pv<<<2304, 256, 0, stream>>>(P, Vt, Oc);
  // waves: 128*12 = 1536 -> 192 blocks
  mha_out<<<192, 256, 0, stream>>>(Oc, woh, out_b, out);
}
